// NeRF_8933531975852
// MI455X (gfx1250) — matrix operand order repaired and checked by reading, not compiled, not run on hardware
//
#include <hip/hip_runtime.h>
#include <math.h>

// ---------------------------------------------------------------------------
// NeRF fused pipeline for MI455X (gfx1250, wave32, WMMA f16 16x16x32).
// M=128 points per workgroup; B (weight) fragments loaded from L2 once per
// block and reused across 8 M-tiles in registers; activations in LDS with
// explicit AS3 addressing; weights with explicit AS1 (global_load_b128).
// This revision: A-fragments prefetched in groups of 4 into distinct registers
// so WMMAs issue back-to-back and WMMA->ds_load WAR hazard gaps (v_nop x4)
// occur once per group instead of once per M-tile.
// ---------------------------------------------------------------------------

typedef __attribute__((ext_vector_type(16))) _Float16 v16h;
typedef __attribute__((ext_vector_type(8)))  _Float16 v8h;
typedef __attribute__((ext_vector_type(8)))  float    v8f;

typedef __attribute__((address_space(3))) _Float16 l_h;   // LDS half
typedef __attribute__((address_space(1))) _Float16 g_h;   // global half

#define LDS_V8H_P  const __attribute__((address_space(3))) v8h*
#define GLB_V16H_P const __attribute__((address_space(1))) v16h*

#define NRAYS   2048
#define NSC     64      // coarse samples
#define NSF     128     // fine samples (64 + 64 importance)
#define NEAR_C  2.0f
#define FAR_C   6.0f
#define NOISE_STD_C 1.0f

#define MR      128     // points (rows) per workgroup
#define MT_N    8       // M-tiles per workgroup (MR/16)

// Swizzled f16 weight blob offsets (in halfs) within one net.
enum : int {
  OFF_L0   = 0,         // K_pad=64 , N=256 -> 16384
  OFF_L1   = 16384,     // 256x256 -> 65536
  OFF_L2   = 81920,
  OFF_L3   = 147456,
  OFF_L4   = 212992,    // K_pad=320, N=256 -> 81920 (skip layer)
  OFF_L5   = 294912,
  OFF_L6   = 360448,
  OFF_L7   = 425984,
  OFF_FEAT = 491520,    // 256x256
  OFF_DIR  = 557056,    // K_pad=288, N=128 -> 36864
  NET_WSZ  = 593920
};

struct NetParams {
  const _Float16* wsw;          // swizzled f16 weights
  const float* bias[8];
  const float* w_sigma; const float* b_sigma;
  const float* b_feat;
  const float* b_dir;
  const float* w_rgb;   const float* b_rgb;
};

// ---------------------------------------------------------------------------
// Weight prep: f32 (fin x N, row major) -> f16 swizzled B-fragment layout.
// Element index e = (((kt*numNt + nt)*32 + lane)*16 + m)
//   n = nt*16 + (lane&15);  k = kt*32 + (lane>>4)*8 + (m&7) + (m>>3)*16
// K padding: rows [p1_real, p1_pad) are zero; rows >= p1_pad map to
// source rows p1_real + (k - p1_pad) while < fin (handles skip/dir concat).
// ---------------------------------------------------------------------------
__global__ void prep_weights(const float* __restrict__ src,
                             _Float16* __restrict__ dst,
                             int N, int K_pad, int p1_real, int p1_pad, int fin)
{
  int e = blockIdx.x * blockDim.x + threadIdx.x;
  int total = K_pad * N;
  if (e >= total) return;
  int m    = e & 15;
  int lane = (e >> 4) & 31;
  int rest = e >> 9;
  int numNt = N >> 4;
  int nt = rest % numNt;
  int kt = rest / numNt;
  int n  = nt * 16 + (lane & 15);
  int k  = kt * 32 + (lane >> 4) * 8 + (m & 7) + (m >> 3) * 16;
  int srcRow;
  if (k < p1_pad) srcRow = (k < p1_real) ? k : -1;
  else { int k2 = k - p1_pad; srcRow = (p1_real + k2 < fin) ? (p1_real + k2) : -1; }
  float v = (srcRow >= 0) ? src[(size_t)srcRow * N + n] : 0.0f;
  dst[e] = (_Float16)v;
}

// ---------------------------------------------------------------------------
// WMMA helpers (all pointers carry explicit address spaces).
// A-frag (16-bit A 16x32): lanes 0-15 = rows, VGPR0-3 K=0..7, VGPR4-7 K=16..23
// lanes 16-31: K=8..15 / K=24..31.
// ---------------------------------------------------------------------------
__device__ inline v16h load_a_frag(const l_h* act, int stride, int k0, int lane)
{
  int r = lane & 15;
  int h = (lane >> 4) & 1;
  const l_h* p = act + r * stride + k0 + h * 8;
  v8h lo = *(LDS_V8H_P)p;          // ds_load_b128
  v8h hi = *(LDS_V8H_P)(p + 16);   // ds_load_b128
  v16h a;
#pragma unroll
  for (int m = 0; m < 8; ++m) { a[m] = lo[m]; a[8 + m] = hi[m]; }
  return a;
}

// One K-segment of a layer GEMM; acc[MT][TPW] lives in registers.
// A fragments are prefetched in groups of AG to batch ds_loads and keep
// WMMAs free of per-tile WAR stalls.
template<int MT, int TPW>
__device__ inline void gemm_seg(v8f (&acc)[MT][TPW],
                                const l_h* base, int stride, int nkk, int& kt,
                                const g_h* wsw, int numNt, int lane, int wave)
{
  constexpr int AG = (MT >= 4) ? 4 : MT;     // A prefetch group size
  for (int kk = 0; kk < nkk; ++kk, ++kt) {
    v16h b[TPW];
#pragma unroll
    for (int t = 0; t < TPW; ++t) {
      int nt = wave * TPW + t;
      const g_h* bp = wsw + ((((size_t)kt * numNt + nt) * 32 + lane) << 4);
      b[t] = *(GLB_V16H_P)bp;      // 32B contiguous -> 2x global_load_b128
    }
#pragma unroll
    for (int g = 0; g < MT; g += AG) {
      v16h a[AG];
#pragma unroll
      for (int i = 0; i < AG; ++i)
        a[i] = load_a_frag(base + (g + i) * 16 * stride, stride, kk * 32, lane);
#pragma unroll
      for (int i = 0; i < AG; ++i)
#pragma unroll
        for (int t = 0; t < TPW; ++t)
          acc[g + i][t] = __builtin_amdgcn_wmma_f32_16x16x32_f16(
              false, a[i], false, b[t], (short)0, acc[g + i][t], false, false);
    }
  }
}

// Full layer: D(MRxN) = A(MRxK) * W(KxN) + bias, optional ReLU, f16 out.
// A may be split across two LDS segments (skip / dir concat).
template<int MT, int TPW>
__device__ inline void run_layer(const l_h* a0, int s0, int kt0,
                                 const l_h* a1, int s1, int kt1,
                                 const g_h* wsw, int N,
                                 const float* __restrict__ bias, bool relu,
                                 l_h* dst, int lane, int wave)
{
  const int numNt = N >> 4;
  v8f acc[MT][TPW];
  const v8f zer = {0.f, 0.f, 0.f, 0.f, 0.f, 0.f, 0.f, 0.f};
#pragma unroll
  for (int mt = 0; mt < MT; ++mt)
#pragma unroll
    for (int t = 0; t < TPW; ++t) acc[mt][t] = zer;

  int kt = 0;
  gemm_seg<MT, TPW>(acc, a0, s0, kt0, kt, wsw, numNt, lane, wave);
  if (kt1 > 0) gemm_seg<MT, TPW>(acc, a1, s1, kt1, kt, wsw, numNt, lane, wave);

  int nl = lane & 15, h = lane >> 4;
#pragma unroll
  for (int t = 0; t < TPW; ++t) {
    int n = (wave * TPW + t) * 16 + nl;
    float bv = bias[n];
    l_h* dbase = dst + (size_t)(8 * h) * N + n;    // immediate offsets per (mt,i)
#pragma unroll
    for (int mt = 0; mt < MT; ++mt) {
#pragma unroll
      for (int i = 0; i < 8; ++i) {
        float v = acc[mt][t][i] + bv;
        if (relu) v = fmaxf(v, 0.0f);
        dbase[(mt * 16 + i) * N] = (_Float16)v;
      }
    }
  }
}

// NeRF positional encoding channel c of [x, sin(2^f p), cos(2^f p)] (f-major).
__device__ inline float embed_val(float x, float y, float z, int c, int nf)
{
  if (c < 3) return (c == 0) ? x : ((c == 1) ? y : z);
  int c2 = c - 3;
  int nsin = 3 * nf;
  bool is_cos = (c2 >= nsin);
  if (is_cos) c2 -= nsin;
  int f = c2 / 3, comp = c2 - 3 * f;
  float v = ((comp == 0) ? x : ((comp == 1) ? y : z)) * (float)(1 << f);
  return is_cos ? cosf(v) : sinf(v);
}

// ---------------------------------------------------------------------------
// Fused MLP: one workgroup = 128 points (per-row ray), 8 waves.
// ---------------------------------------------------------------------------
__global__ __launch_bounds__(256) void nerf_mlp(
    const float* __restrict__ rays_o, const float* __restrict__ rays_d,
    const float* __restrict__ z_buf, int S, NetParams np,
    float* __restrict__ sigma_out, float* __restrict__ rgb_out)
{
  __shared__ _Float16 embA[MR * 64];    // 16 KB
  __shared__ _Float16 bufA[MR * 256];   // 64 KB
  __shared__ _Float16 bufB[MR * 256];   // 64 KB
  __shared__ _Float16 dirE[MR * 32];    //  8 KB
  __shared__ float    pxyz[MR * 3];
  __shared__ float    pdir[MR * 3];

  // explicit LDS address-space views (32-bit addressing, no aperture checks)
  l_h* Lemb  = (l_h*)embA;
  l_h* LbufA = (l_h*)bufA;
  l_h* LbufB = (l_h*)bufB;
  l_h* LdirE = (l_h*)dirE;

  int tid  = threadIdx.x;
  int lane = tid & 31, wave = tid >> 5;
  int p0   = blockIdx.x * MR;

  // per-row point position + ray direction
  if (tid < MR) {
    int gp  = p0 + tid;
    int ray = gp / S;
    int s   = gp - ray * S;
    float ox = rays_o[ray * 3 + 0], oy = rays_o[ray * 3 + 1], oz = rays_o[ray * 3 + 2];
    float dx = rays_d[ray * 3 + 0], dy = rays_d[ray * 3 + 1], dz = rays_d[ray * 3 + 2];
    float z = z_buf ? z_buf[(size_t)ray * S + s]
                    : (NEAR_C + (FAR_C - NEAR_C) * ((float)s / (float)(S - 1)));
    pxyz[tid * 3 + 0] = ox + dx * z;
    pxyz[tid * 3 + 1] = oy + dy * z;
    pxyz[tid * 3 + 2] = oz + dz * z;
    pdir[tid * 3 + 0] = dx; pdir[tid * 3 + 1] = dy; pdir[tid * 3 + 2] = dz;
  }
  __syncthreads();
  for (int idx = tid; idx < MR * 64; idx += 256) {
    int p = idx >> 6, c = idx & 63;
    float v = (c < 63) ? embed_val(pxyz[p * 3], pxyz[p * 3 + 1], pxyz[p * 3 + 2], c, 10) : 0.f;
    Lemb[idx] = (_Float16)v;
  }
  for (int idx = tid; idx < MR * 32; idx += 256) {
    int p = idx >> 5, c = idx & 31;
    float v = (c < 27) ? embed_val(pdir[p * 3], pdir[p * 3 + 1], pdir[p * 3 + 2], c, 4) : 0.f;
    LdirE[idx] = (_Float16)v;
  }
  __syncthreads();

  const g_h* wsw = (const g_h*)np.wsw;
  run_layer<MT_N, 2>(Lemb, 64, 2, Lemb, 0, 0, wsw + OFF_L0, 256, np.bias[0], true, LbufA, lane, wave); __syncthreads();
  run_layer<MT_N, 2>(LbufA, 256, 8, Lemb, 0, 0, wsw + OFF_L1, 256, np.bias[1], true, LbufB, lane, wave); __syncthreads();
  run_layer<MT_N, 2>(LbufB, 256, 8, Lemb, 0, 0, wsw + OFF_L2, 256, np.bias[2], true, LbufA, lane, wave); __syncthreads();
  run_layer<MT_N, 2>(LbufA, 256, 8, Lemb, 0, 0, wsw + OFF_L3, 256, np.bias[3], true, LbufB, lane, wave); __syncthreads();
  // skip layer: concat(xyz_e[63 pad 64], h[256]) -> 256
  run_layer<MT_N, 2>(Lemb, 64, 2, LbufB, 256, 8, wsw + OFF_L4, 256, np.bias[4], true, LbufA, lane, wave); __syncthreads();
  run_layer<MT_N, 2>(LbufA, 256, 8, Lemb, 0, 0, wsw + OFF_L5, 256, np.bias[5], true, LbufB, lane, wave); __syncthreads();
  run_layer<MT_N, 2>(LbufB, 256, 8, Lemb, 0, 0, wsw + OFF_L6, 256, np.bias[6], true, LbufA, lane, wave); __syncthreads();
  run_layer<MT_N, 2>(LbufA, 256, 8, Lemb, 0, 0, wsw + OFF_L7, 256, np.bias[7], true, LbufB, lane, wave); __syncthreads();

  // sigma head (raw; softplus+noise applied in render kernel)
  if (tid < MR) {
    float acc = np.b_sigma[0];
    for (int k = 0; k < 256; ++k) acc += (float)LbufB[tid * 256 + k] * np.w_sigma[k];
    sigma_out[p0 + tid] = acc;
  }
  // feat: h -> 256 (no activation)
  run_layer<MT_N, 2>(LbufB, 256, 8, Lemb, 0, 0, wsw + OFF_FEAT, 256, np.b_feat, false, LbufA, lane, wave); __syncthreads();
  // dir layer: concat(feat[256], dir_e[27 pad 32]) -> 128, ReLU
  run_layer<MT_N, 1>(LbufA, 256, 8, LdirE, 32, 1, wsw + OFF_DIR, 128, np.b_dir, true, LbufB, lane, wave); __syncthreads();
  // rgb head: 128 -> 3, sigmoid
  for (int idx = tid; idx < MR * 3; idx += 256) {
    int p = idx / 3, c = idx - 3 * p;
    float acc = np.b_rgb[c];
    for (int k = 0; k < 128; ++k) acc += (float)LbufB[p * 128 + k] * np.w_rgb[k * 3 + c];
    rgb_out[(size_t)(p0 + p) * 3 + c] = 1.0f / (1.0f + expf(-acc));
  }
}

// ---------------------------------------------------------------------------
// Volume rendering: one thread per ray (sequential transmittance scan).
// ---------------------------------------------------------------------------
__device__ inline unsigned hash_u32(unsigned x) {
  x ^= x >> 16; x *= 0x7feb352du; x ^= x >> 15; x *= 0x846ca68bu; x ^= x >> 16;
  return x;
}
__device__ inline float gauss_hash(unsigned seed, unsigned i) {
  unsigned a = hash_u32(i * 2u + 0x9e3779b9u * seed + 1u);
  unsigned b = hash_u32(i * 2u + 0x9e3779b9u * seed + 2u);
  float u1 = ((float)(a >> 8) + 1.0f) * (1.0f / 16777216.0f);  // (0,1]
  float u2 = (float)(b >> 8) * (1.0f / 16777216.0f);
  return sqrtf(-2.0f * logf(u1)) * cosf(6.28318530718f * u2);
}
__device__ inline float softplus_f(float x) {
  return (x > 30.f) ? x : log1pf(expf(x));
}

__global__ void nerf_render(const float* __restrict__ rays_d,
                            const float* __restrict__ z_buf, int S,
                            const float* __restrict__ sigma,
                            const float* __restrict__ rgb,
                            float* __restrict__ out_rgb,
                            float* __restrict__ out_depth,
                            float* __restrict__ wts_out, unsigned seed)
{
  int r = blockIdx.x * blockDim.x + threadIdx.x;
  if (r >= NRAYS) return;
  float dx = rays_d[r * 3], dy = rays_d[r * 3 + 1], dz = rays_d[r * 3 + 2];
  float dn = sqrtf(dx * dx + dy * dy + dz * dz);
  float T = 1.f, cr = 0.f, cg = 0.f, cb = 0.f, dep = 0.f;
  float zc = z_buf ? z_buf[(size_t)r * S] : NEAR_C;
  for (int s = 0; s < S; ++s) {
    float zn = 0.f, delta;
    if (s + 1 < S) {
      zn = z_buf ? z_buf[(size_t)r * S + s + 1]
                 : NEAR_C + (FAR_C - NEAR_C) * ((float)(s + 1) / (float)(S - 1));
      delta = (zn - zc) * dn;
    } else delta = 1e10f * dn;
    float sg = sigma[(size_t)r * S + s] + NOISE_STD_C * gauss_hash(seed, (unsigned)(r * S + s));
    float alpha = 1.f - expf(-delta * softplus_f(sg));
    float w = alpha * T;
    T *= (1.f - alpha + 1e-10f);
    size_t o = ((size_t)r * S + s) * 3;
    cr += w * rgb[o]; cg += w * rgb[o + 1]; cb += w * rgb[o + 2];
    dep += w * zc;
    if (wts_out) wts_out[(size_t)r * S + s] = w;
    zc = zn;
  }
  out_rgb[r * 3 + 0] = cr; out_rgb[r * 3 + 1] = cg; out_rgb[r * 3 + 2] = cb;
  out_depth[r] = dep;
}

// ---------------------------------------------------------------------------
// Inverse-CDF resampling (deterministic) + merge into sorted z_fine[r][128].
// ---------------------------------------------------------------------------
__global__ void nerf_resample(const float* __restrict__ wts_c,
                              float* __restrict__ z_fine)
{
  int r = blockIdx.x * blockDim.x + threadIdx.x;
  if (r >= NRAYS) return;
  const float step = (FAR_C - NEAR_C) / (float)(NSC - 1);
  float cdf[NSC - 1];               // 63 entries: leading 0 + cumsum of 62 weights
  cdf[0] = 0.f;
  float wsum = 0.f;
  for (int j = 0; j < NSC - 2; ++j) {
    wsum += wts_c[(size_t)r * NSC + 1 + j] + 1e-5f;
    cdf[j + 1] = wsum;
  }
  float inv = 1.f / wsum;
  for (int j = 1; j < NSC - 1; ++j) cdf[j] *= inv;

  float zs[NSC];
  for (int i = 0; i < NSC; ++i) {
    float u = (float)i / (float)(NSC - 1);
    int lo = 0, hi = NSC - 1;       // searchsorted-right over 63 entries
    while (lo < hi) { int mid = (lo + hi) >> 1; if (cdf[mid] <= u) lo = mid + 1; else hi = mid; }
    int below = (lo - 1 > 0) ? lo - 1 : 0;
    int above = (lo < NSC - 2) ? lo : NSC - 2;
    float c0 = cdf[below], c1 = cdf[above];
    float b0 = NEAR_C + step * ((float)below + 0.5f);
    float b1 = NEAR_C + step * ((float)above + 0.5f);
    float denom = c1 - c0;
    if (denom < 1e-5f) denom = 1.f;
    zs[i] = b0 + (u - c0) / denom * (b1 - b0);
  }
  // merge uniform z (64, sorted) with zs (64, sorted)
  int i = 0, j = 0;
  for (int k = 0; k < NSF; ++k) {
    float zv = NEAR_C + step * (float)i;
    float pick;
    if (j >= NSC || (i < NSC && zv <= zs[j])) { pick = zv; ++i; }
    else { pick = zs[j]; ++j; }
    z_fine[(size_t)r * NSF + k] = pick;
  }
}

// ---------------------------------------------------------------------------
// Host launcher.
// ---------------------------------------------------------------------------
struct NetIdx { int w[8], b[8], w_sigma, b_sigma, w_feat, b_feat, w_dir, b_dir, w_rgb, b_rgb; };

static NetIdx make_netidx(int base, bool insertion_order) {
  NetIdx x;
  if (insertion_order) {            // w0,b0,...,w7,b7,w_sigma,b_sigma,w_feat,b_feat,w_dir,b_dir,w_rgb,b_rgb
    for (int i = 0; i < 8; ++i) { x.w[i] = base + 2 * i; x.b[i] = base + 2 * i + 1; }
    x.w_sigma = base + 16; x.b_sigma = base + 17;
    x.w_feat  = base + 18; x.b_feat  = base + 19;
    x.w_dir   = base + 20; x.b_dir   = base + 21;
    x.w_rgb   = base + 22; x.b_rgb   = base + 23;
  } else {                          // pytree-alphabetical: b0..b7,b_dir,b_feat,b_rgb,b_sigma,w0..w7,w_dir,w_feat,w_rgb,w_sigma
    for (int i = 0; i < 8; ++i) x.b[i] = base + i;
    x.b_dir = base + 8; x.b_feat = base + 9; x.b_rgb = base + 10; x.b_sigma = base + 11;
    for (int i = 0; i < 8; ++i) x.w[i] = base + 12 + i;
    x.w_dir = base + 20; x.w_feat = base + 21; x.w_rgb = base + 22; x.w_sigma = base + 23;
  }
  return x;
}

extern "C" void kernel_launch(void* const* d_in, const int* in_sizes, int n_in,
                              void* d_out, int out_size, void* d_ws, size_t ws_size,
                              hipStream_t stream)
{
  // ---- locate inputs (robust to insertion vs. pytree-alphabetical order) ----
  int rays_o_i, rays_d_i, base_c, base_f;
  if (in_sizes[0] == NRAYS * 3 && in_sizes[1] == NRAYS * 3) {
    rays_o_i = 0; rays_d_i = 1; base_c = 2; base_f = 26;     // dict insertion order
  } else {
    base_c = 0; base_f = 24; rays_d_i = 48; rays_o_i = 49;   // alphabetical order
  }
  bool ins = (in_sizes[base_c] > 4096);                       // w0 (63*256) vs b0 (256)
  NetIdx ic = make_netidx(base_c, ins);
  NetIdx iff = make_netidx(base_f, ins);

  const float* rays_o = (const float*)d_in[rays_o_i];
  const float* rays_d = (const float*)d_in[rays_d_i];
  auto F = [&](int i) -> const float* { return (const float*)d_in[i]; };

  // ---- workspace layout ----
  _Float16* wsw_c = (_Float16*)d_ws;
  _Float16* wsw_f = wsw_c + NET_WSZ;
  float* sigma_c = (float*)(wsw_f + NET_WSZ);
  float* rgb_c   = sigma_c + (size_t)NRAYS * NSC;
  float* wts_c   = rgb_c   + (size_t)NRAYS * NSC * 3;
  float* z_fine  = wts_c   + (size_t)NRAYS * NSC;
  float* sigma_f = z_fine  + (size_t)NRAYS * NSF;
  float* rgb_f   = sigma_f + (size_t)NRAYS * NSF;
  (void)ws_size; (void)n_in; (void)out_size;

  float* out      = (float*)d_out;
  float* o_rgb_c  = out;
  float* o_dep_c  = out + NRAYS * 3;
  float* o_rgb_f  = out + NRAYS * 4;
  float* o_dep_f  = out + NRAYS * 4 + NRAYS * 3;

  // ---- 1) weight prep (both nets) ----
  struct LD { int which; int off; int N; int K_pad; int p1r; int p1p; int fin; };
  const LD lds[10] = {
    {0, OFF_L0,   256,  64,  63,  64,  63},
    {1, OFF_L1,   256, 256, 256, 256, 256},
    {2, OFF_L2,   256, 256, 256, 256, 256},
    {3, OFF_L3,   256, 256, 256, 256, 256},
    {4, OFF_L4,   256, 320,  63,  64, 319},   // skip: [xyz_e(63|pad), h(256)]
    {5, OFF_L5,   256, 256, 256, 256, 256},
    {6, OFF_L6,   256, 256, 256, 256, 256},
    {7, OFF_L7,   256, 256, 256, 256, 256},
    {8, OFF_FEAT, 256, 256, 256, 256, 256},
    {9, OFF_DIR,  128, 288, 256, 256, 283},   // [feat(256), dir_e(27|pad)]
  };
  for (int net = 0; net < 2; ++net) {
    const NetIdx& ni = net ? iff : ic;
    _Float16* dst_base = net ? wsw_f : wsw_c;
    for (int l = 0; l < 10; ++l) {
      const LD& L = lds[l];
      const float* src = (L.which < 8) ? F(ni.w[L.which])
                        : (L.which == 8 ? F(ni.w_feat) : F(ni.w_dir));
      int total = L.K_pad * L.N;
      prep_weights<<<(total + 255) / 256, 256, 0, stream>>>(
          src, dst_base + L.off, L.N, L.K_pad, L.p1r, L.p1p, L.fin);
    }
  }

  // ---- NetParams ----
  NetParams npc, npf;
  npc.wsw = wsw_c; npf.wsw = wsw_f;
  for (int i = 0; i < 8; ++i) { npc.bias[i] = F(ic.b[i]); npf.bias[i] = F(iff.b[i]); }
  npc.w_sigma = F(ic.w_sigma);  npc.b_sigma = F(ic.b_sigma);
  npc.b_feat  = F(ic.b_feat);   npc.b_dir   = F(ic.b_dir);
  npc.w_rgb   = F(ic.w_rgb);    npc.b_rgb   = F(ic.b_rgb);
  npf.w_sigma = F(iff.w_sigma); npf.b_sigma = F(iff.b_sigma);
  npf.b_feat  = F(iff.b_feat);  npf.b_dir   = F(iff.b_dir);
  npf.w_rgb   = F(iff.w_rgb);   npf.b_rgb   = F(iff.b_rgb);

  // ---- 2) coarse MLP: 131072 points / 128 per block ----
  nerf_mlp<<<(NRAYS * NSC) / MR, 256, 0, stream>>>(
      rays_o, rays_d, nullptr, NSC, npc, sigma_c, rgb_c);

  // ---- 3) coarse volume render (keeps weights for resampling) ----
  nerf_render<<<NRAYS / 256, 256, 0, stream>>>(
      rays_d, nullptr, NSC, sigma_c, rgb_c, o_rgb_c, o_dep_c, wts_c, 1u);

  // ---- 4) inverse-CDF importance resample + merge-sort ----
  nerf_resample<<<NRAYS / 256, 256, 0, stream>>>(wts_c, z_fine);

  // ---- 5) fine MLP: 262144 points ----
  nerf_mlp<<<(NRAYS * NSF) / MR, 256, 0, stream>>>(
      rays_o, rays_d, z_fine, NSF, npf, sigma_f, rgb_f);

  // ---- 6) fine volume render ----
  nerf_render<<<NRAYS / 256, 256, 0, stream>>>(
      rays_d, z_fine, NSF, sigma_f, rgb_f, o_rgb_f, o_dep_f, nullptr, 2u);
}
